// SingularMoELinear_48352741818884
// MI455X (gfx1250) — matrix operand order repaired and checked by reading, not compiled, not run on hardware
//
#include <hip/hip_runtime.h>
#include <hip/hip_bf16.h>
#include <math.h>

// ---------------------------------------------------------------------------
// Problem constants: B=2, S=2048, D=2048, O=2048, E=8, GK=16, K=32, top_k=2.
// N = B*S = 4096.
// ---------------------------------------------------------------------------
#define N_TOK   4096
#define DIM_D   2048
#define DIM_O   2048
#define N_EXP   8
#define GATE_K  16
#define RANK_K  32
#define PROJ_N  (N_EXP * GATE_K)           // 128
#define HK_N    (N_EXP * RANK_K)           // 256
#define KK_AUG  (HK_N + N_EXP)             // 264
#define KK_PAD  288                        // padded to multiple of BK=32

#define LDS_STRIDE 40                      // 32 + 8 bf16 pad (bank-conflict free)

typedef __attribute__((ext_vector_type(16))) __bf16 v16bf;
typedef __attribute__((ext_vector_type(8)))  float  v8f;
typedef int v4i_gcc __attribute__((vector_size(16)));   // matches builtin param

#define GPTR(p) ((__attribute__((address_space(1))) v4i_gcc*)(p))
#define LPTR(p) ((__attribute__((address_space(3))) v4i_gcc*)(p))

#if __has_builtin(__builtin_amdgcn_global_load_async_to_lds_b128)
#define HAVE_ASYNC_LDS 1
#else
#define HAVE_ASYNC_LDS 0
#endif

__device__ __forceinline__ void wait_async0() {
#if __has_builtin(__builtin_amdgcn_s_wait_asynccnt)
    __builtin_amdgcn_s_wait_asynccnt(0);
#else
    asm volatile("s_wait_asynccnt 0x0" ::: "memory");
#endif
}

union FragU { v16bf v; uint4 q[2]; };

__device__ __forceinline__ void split_bf16(float x, __bf16& hi, __bf16& lo) {
    hi = (__bf16)x;
    lo = (__bf16)(x - (float)hi);
}

// ---------------------------------------------------------------------------
// One-time fp32 -> bf16 hi/lo plane split (elementwise, float4 vectorized).
// ---------------------------------------------------------------------------
__global__ void split_kernel(const float* __restrict__ src,
                             __bf16* __restrict__ hi, __bf16* __restrict__ lo,
                             int n) {
    const int i = (blockIdx.x * blockDim.x + threadIdx.x) * 4;
    if (i >= n) return;
    const float4 v = *(const float4*)(src + i);
    __bf16 h, l;
    split_bf16(v.x, h, l); hi[i + 0] = h; lo[i + 0] = l;
    split_bf16(v.y, h, l); hi[i + 1] = h; lo[i + 1] = l;
    split_bf16(v.z, h, l); hi[i + 2] = h; lo[i + 2] = l;
    split_bf16(v.w, h, l); hi[i + 3] = h; lo[i + 3] = l;
}

// A-fragment (16x32 bf16, M x K) per ISA 7.12.2:
//   lanes 0-15 : row M=lane,    K = {0..7, 16..23}
//   lanes 16-31: row M=lane-16, K = {8..15, 24..31}
__device__ __forceinline__ v16bf load_frag_A(const __bf16* tile, int m0, int lane) {
    const int l   = lane & 15;
    const int sel = lane >> 4;
    const __bf16* p = tile + (m0 + l) * LDS_STRIDE + sel * 8;
    FragU f;
    f.q[0] = *(const uint4*)(p);
    f.q[1] = *(const uint4*)(p + 16);
    return f.v;
}

// B-fragment (32x16 bf16, K x N), sourced from row-major W[o, d]:
//   lanes 0-15 : col N=lane,    K = {0..7, 16..23}
//   lanes 16-31: col N=lane-16, K = {8..15, 24..31}
__device__ __forceinline__ v16bf load_frag_B(const __bf16* tile, int n0, int lane) {
    const int l   = lane & 15;
    const int sel = lane >> 4;
    const __bf16* p = tile + (n0 + l) * LDS_STRIDE + sel * 8;
    FragU f;
    f.q[0] = *(const uint4*)(p);
    f.q[1] = *(const uint4*)(p + 16);
    return f.v;
}

// Issue one 128x32 tile pair (A and B, hi+lo planes) into an LDS buffer.
// 512 threads: thread t handles row t/4, cols (t&3)*8 .. +7 (16 B) per plane.
// Preferred path: GLOBAL_LOAD_ASYNC_TO_LDS_B128 (no VGPR staging, ASYNCcnt).
__device__ __forceinline__ void issue_tile(const __bf16* __restrict__ Ah,
                                           const __bf16* __restrict__ Al,
                                           const __bf16* __restrict__ Bh,
                                           const __bf16* __restrict__ Bl,
                                           __bf16* dAh, __bf16* dAl,
                                           __bf16* dBh, __bf16* dBl,
                                           int m0, int n0, int kt, int Kd,
                                           int tidx) {
    const int r = tidx >> 2;
    const int c = (tidx & 3) * 8;
    const size_t ao = (size_t)(m0 + r) * Kd + kt + c;
    const size_t bo = (size_t)(n0 + r) * Kd + kt + c;
    const int o = r * LDS_STRIDE + c;
#if HAVE_ASYNC_LDS
    __builtin_amdgcn_global_load_async_to_lds_b128(GPTR(Ah + ao), LPTR(dAh + o), 0, 0);
    __builtin_amdgcn_global_load_async_to_lds_b128(GPTR(Al + ao), LPTR(dAl + o), 0, 0);
    __builtin_amdgcn_global_load_async_to_lds_b128(GPTR(Bh + bo), LPTR(dBh + o), 0, 0);
    __builtin_amdgcn_global_load_async_to_lds_b128(GPTR(Bl + bo), LPTR(dBl + o), 0, 0);
#else
    // Fallback: short-lived register staging (only 16 VGPRs live).
    uint4 a0 = *(const uint4*)(Ah + ao);
    uint4 a1 = *(const uint4*)(Al + ao);
    uint4 b0 = *(const uint4*)(Bh + bo);
    uint4 b1 = *(const uint4*)(Bl + bo);
    *(uint4*)(dAh + o) = a0;
    *(uint4*)(dAl + o) = a1;
    *(uint4*)(dBh + o) = b0;
    *(uint4*)(dBl + o) = b1;
#endif
}

// ---------------------------------------------------------------------------
// Split-precision GEMM on pre-split bf16 planes:
//   C[M,Nout] (+)= A[M,Kd] * Bw[Nout,Kd]^T (+bias),  C ~= Ah*Bh + Ah*Bl + Al*Bh
// M%128==0, Nout%128==0, Kd%32==0.
// 512 threads = 16 waves; block tile 128x128; wave tile 32x32.
// Double-buffered LDS fed by async global->LDS copies.
// ---------------------------------------------------------------------------
template <bool ADD_BIAS, bool ACCUM>
__global__ __launch_bounds__(512)
void gemm_planes(const __bf16* __restrict__ Ah, const __bf16* __restrict__ Al,
                 const __bf16* __restrict__ Bh, const __bf16* __restrict__ Bl,
                 const float* __restrict__ bias, float* __restrict__ C,
                 int M, int Nout, int Kd) {
    __shared__ __align__(16) __bf16 sAh[2][128 * LDS_STRIDE];
    __shared__ __align__(16) __bf16 sAl[2][128 * LDS_STRIDE];
    __shared__ __align__(16) __bf16 sBh[2][128 * LDS_STRIDE];
    __shared__ __align__(16) __bf16 sBl[2][128 * LDS_STRIDE];

    const int tidx = threadIdx.x;
    const int lane = tidx & 31;
    const int wave = tidx >> 5;            // 16 waves
    const int wm   = (wave & 3) * 32;      // 4 waves along M -> 32 rows
    const int wn   = (wave >> 2) * 32;     // 4 waves along N -> 32 cols
    const int m0   = blockIdx.y * 128;
    const int n0   = blockIdx.x * 128;

    v8f acc[2][2] = {};

    issue_tile(Ah, Al, Bh, Bl, sAh[0], sAl[0], sBh[0], sBl[0],
               m0, n0, 0, Kd, tidx);
    wait_async0();
    __syncthreads();

    int cur = 0;
    for (int kt = 0; kt < Kd; kt += 32) {
        if (kt + 32 < Kd) {
            issue_tile(Ah, Al, Bh, Bl, sAh[cur ^ 1], sAl[cur ^ 1],
                       sBh[cur ^ 1], sBl[cur ^ 1], m0, n0, kt + 32, Kd, tidx);
            if (kt + 64 < Kd) {                 // global_prefetch_b8
                const int r = tidx >> 2;
                __builtin_prefetch(Ah + (size_t)(m0 + r) * Kd + kt + 64, 0, 3);
                __builtin_prefetch(Bh + (size_t)(n0 + r) * Kd + kt + 64, 0, 3);
            }
        }

        // WMMA on current buffer while async copies fill the other one.
        v16bf a_hi[2], a_lo[2];
        #pragma unroll
        for (int i = 0; i < 2; ++i) {
            a_hi[i] = load_frag_A(sAh[cur], wm + 16 * i, lane);
            a_lo[i] = load_frag_A(sAl[cur], wm + 16 * i, lane);
        }
        #pragma unroll
        for (int j = 0; j < 2; ++j) {
            const v16bf b_hi = load_frag_B(sBh[cur], wn + 16 * j, lane);
            const v16bf b_lo = load_frag_B(sBl[cur], wn + 16 * j, lane);
            #pragma unroll
            for (int i = 0; i < 2; ++i) {
                acc[i][j] = __builtin_amdgcn_wmma_f32_16x16x32_bf16(
                    false, a_hi[i], false, b_hi, (short)0, acc[i][j], false, false);
                acc[i][j] = __builtin_amdgcn_wmma_f32_16x16x32_bf16(
                    false, a_hi[i], false, b_lo, (short)0, acc[i][j], false, false);
                acc[i][j] = __builtin_amdgcn_wmma_f32_16x16x32_bf16(
                    false, a_lo[i], false, b_hi, (short)0, acc[i][j], false, false);
            }
        }

        wait_async0();       // next buffer fully in LDS
        __syncthreads();     // + all waves done reading current buffer
        cur ^= 1;
    }

    // Epilogue: elem r of v8f -> row = base + r + (lane>=16)*8, col = base+(lane&15)
    const int rsel = (lane >> 4) * 8;
    const int csel = lane & 15;
    #pragma unroll
    for (int i = 0; i < 2; ++i) {
        const int rowbase = m0 + wm + 16 * i + rsel;
        #pragma unroll
        for (int j = 0; j < 2; ++j) {
            const int col = n0 + wn + 16 * j + csel;
            #pragma unroll
            for (int r = 0; r < 8; ++r) {
                const size_t off = (size_t)(rowbase + r) * Nout + col;
                float v = acc[i][j][r];
                if (ADD_BIAS) v += bias[col];
                if (ACCUM) C[off] += v; else C[off] = v;
            }
        }
    }
}

// ---------------------------------------------------------------------------
// Pack  U2[o, e*32+k] = u[e,o,k];  U2[o, 256+e] = expert_bias[e,o];  pad = 0.
// Emits bf16 hi/lo planes directly.
// ---------------------------------------------------------------------------
__global__ void pack_u_kernel(const float* __restrict__ u,
                              const float* __restrict__ expert_bias,
                              __bf16* __restrict__ U2h, __bf16* __restrict__ U2l) {
    const int idx = blockIdx.x * blockDim.x + threadIdx.x;   // o * KK_PAD + kk
    if (idx >= DIM_O * KK_PAD) return;
    const int o  = idx / KK_PAD;
    const int kk = idx - o * KK_PAD;
    float v = 0.0f;
    if (kk < HK_N) {
        const int e = kk >> 5, k = kk & 31;
        v = u[((size_t)e * DIM_O + o) * RANK_K + k];
    } else if (kk < KK_AUG) {
        v = expert_bias[(size_t)(kk - HK_N) * DIM_O + o];
    }
    __bf16 h, l;
    split_bf16(v, h, l);
    U2h[idx] = h; U2l[idx] = l;
}

// ---------------------------------------------------------------------------
// Router: logits -> softmax -> top-2 -> normalized dense weights [N, E].
// ---------------------------------------------------------------------------
__global__ void router_kernel(const float* __restrict__ proj,
                              float* __restrict__ wd) {
    const int n = blockIdx.x * blockDim.x + threadIdx.x;
    if (n >= N_TOK) return;
    const float* p = proj + (size_t)n * PROJ_N;
    float logit[N_EXP];
    float mx = -1e30f;
    #pragma unroll
    for (int e = 0; e < N_EXP; ++e) {
        float s = 0.0f;
        #pragma unroll
        for (int g = 0; g < GATE_K; ++g) { float t = p[e * GATE_K + g]; s += t * t; }
        logit[e] = sqrtf(s);
        mx = fmaxf(mx, logit[e]);
    }
    float w[N_EXP], sum = 0.0f;
    #pragma unroll
    for (int e = 0; e < N_EXP; ++e) { w[e] = expf(logit[e] - mx); sum += w[e]; }
    const float inv = 1.0f / sum;
    #pragma unroll
    for (int e = 0; e < N_EXP; ++e) w[e] *= inv;
    int i1 = 0;
    #pragma unroll
    for (int e = 1; e < N_EXP; ++e) if (w[e] > w[i1]) i1 = e;
    int i2 = (i1 == 0) ? 1 : 0;
    #pragma unroll
    for (int e = 0; e < N_EXP; ++e) if (e != i1 && w[e] > w[i2]) i2 = e;
    const float norm = 1.0f / (w[i1] + w[i2]);
    #pragma unroll
    for (int e = 0; e < N_EXP; ++e)
        wd[(size_t)n * N_EXP + e] =
            (e == i1) ? w[i1] * norm : ((e == i2) ? w[i2] * norm : 0.0f);
}

// ---------------------------------------------------------------------------
// hkw_aug[n, e*32+k] = hk[n, e*32+k] * wd[n,e];  hkw_aug[n, 256+e] = wd[n,e];
// pad = 0.  Emits bf16 hi/lo planes directly.
// ---------------------------------------------------------------------------
__global__ void scale_hk_kernel(const float* __restrict__ hk,
                                const float* __restrict__ wd,
                                __bf16* __restrict__ Hh, __bf16* __restrict__ Hl) {
    const int idx = blockIdx.x * blockDim.x + threadIdx.x;   // n * KK_PAD + kk
    if (idx >= N_TOK * KK_PAD) return;
    const int n  = idx / KK_PAD;
    const int kk = idx - n * KK_PAD;
    float v = 0.0f;
    if (kk < HK_N) {
        v = hk[(size_t)n * HK_N + kk] * wd[(size_t)n * N_EXP + (kk >> 5)];
    } else if (kk < KK_AUG) {
        v = wd[(size_t)n * N_EXP + (kk - HK_N)];
    }
    __bf16 h, l;
    split_bf16(v, h, l);
    Hh[idx] = h; Hl[idx] = l;
}

// ---------------------------------------------------------------------------
// Host orchestration.
// Inputs: hidden_states, W, b, gate_w, u, svh, expert_bias, top_k
// ---------------------------------------------------------------------------
extern "C" void kernel_launch(void* const* d_in, const int* in_sizes, int n_in,
                              void* d_out, int out_size, void* d_ws, size_t ws_size,
                              hipStream_t stream) {
    (void)in_sizes; (void)n_in; (void)out_size; (void)ws_size;
    const float* x      = (const float*)d_in[0];   // [N, D]
    const float* W      = (const float*)d_in[1];   // [O, D]
    const float* b      = (const float*)d_in[2];   // [O]
    const float* gate_w = (const float*)d_in[3];   // [128, D]
    const float* u      = (const float*)d_in[4];   // [E, O, K]
    const float* svh    = (const float*)d_in[5];   // [256, D]
    const float* ebias  = (const float*)d_in[6];   // [E, O]
    float* out = (float*)d_out;                    // [N, O]

    // Workspace layout
    char* ws = (char*)d_ws;
    float* proj = (float*)ws;                               ws += (size_t)N_TOK * PROJ_N * 4;
    float* wd   = (float*)ws;                               ws += (size_t)N_TOK * N_EXP  * 4;
    float* hk   = (float*)ws;                               ws += (size_t)N_TOK * HK_N   * 4;
    __bf16* Xh  = (__bf16*)ws;                              ws += (size_t)N_TOK * DIM_D  * 2;
    __bf16* Xl  = (__bf16*)ws;                              ws += (size_t)N_TOK * DIM_D  * 2;
    __bf16* Wh  = (__bf16*)ws;                              ws += (size_t)DIM_O * DIM_D  * 2;
    __bf16* Wl  = (__bf16*)ws;                              ws += (size_t)DIM_O * DIM_D  * 2;
    __bf16* Gh  = (__bf16*)ws;                              ws += (size_t)PROJ_N * DIM_D * 2;
    __bf16* Gl  = (__bf16*)ws;                              ws += (size_t)PROJ_N * DIM_D * 2;
    __bf16* Sh  = (__bf16*)ws;                              ws += (size_t)HK_N * DIM_D   * 2;
    __bf16* Sl  = (__bf16*)ws;                              ws += (size_t)HK_N * DIM_D   * 2;
    __bf16* Hh  = (__bf16*)ws;                              ws += (size_t)N_TOK * KK_PAD * 2;
    __bf16* Hl  = (__bf16*)ws;                              ws += (size_t)N_TOK * KK_PAD * 2;
    __bf16* U2h = (__bf16*)ws;                              ws += (size_t)DIM_O * KK_PAD * 2;
    __bf16* U2l = (__bf16*)ws;

    const dim3 blk(256);
    const dim3 gblk(512);
    auto splits = [&](const float* s, __bf16* h, __bf16* l, int n) {
        split_kernel<<<dim3((n / 4 + 255) / 256), blk, 0, stream>>>(s, h, l, n);
    };

    // 1. One-time hi/lo plane splits (streaming, ~50 MB total -> trivial).
    splits(x,      Xh, Xl, N_TOK * DIM_D);
    splits(W,      Wh, Wl, DIM_O * DIM_D);
    splits(gate_w, Gh, Gl, PROJ_N * DIM_D);
    splits(svh,    Sh, Sl, HK_N * DIM_D);

    // 2. Pack expert up-projection + bias into bf16 [O, KK_PAD] planes.
    pack_u_kernel<<<dim3((DIM_O * KK_PAD + 255) / 256), blk, 0, stream>>>(u, ebias, U2h, U2l);

    // 3. Router projection: proj = x @ gate_w^T   [4096 x 128]
    gemm_planes<false, false><<<dim3(PROJ_N / 128, N_TOK / 128), gblk, 0, stream>>>(
        Xh, Xl, Gh, Gl, nullptr, proj, N_TOK, PROJ_N, DIM_D);

    // 4. Softmax / top-2 routing weights.
    router_kernel<<<dim3(N_TOK / 256), blk, 0, stream>>>(proj, wd);

    // 5. Expert down-projection: hk = x @ svh^T   [4096 x 256]
    gemm_planes<false, false><<<dim3(HK_N / 128, N_TOK / 128), gblk, 0, stream>>>(
        Xh, Xl, Sh, Sl, nullptr, hk, N_TOK, HK_N, DIM_D);

    // 6. Scale hk by routing weights; append w_dense columns (expert bias).
    scale_hk_kernel<<<dim3((N_TOK * KK_PAD + 255) / 256), blk, 0, stream>>>(hk, wd, Hh, Hl);

    // 7. Pretrained dense: out = x @ W^T + b      [4096 x 2048]   (dominant)
    gemm_planes<true, false><<<dim3(DIM_O / 128, N_TOK / 128), gblk, 0, stream>>>(
        Xh, Xl, Wh, Wl, b, out, N_TOK, DIM_O, DIM_D);

    // 8. MoE combine: out += hkw_aug @ U2^T       [4096 x 2048], K = 288
    gemm_planes<false, true><<<dim3(DIM_O / 128, N_TOK / 128), gblk, 0, stream>>>(
        Hh, Hl, U2h, U2l, nullptr, out, N_TOK, DIM_O, KK_PAD);
}
